// TransitionUp_420906795557
// MI455X (gfx1250) — matrix operand order repaired and checked by reading, not compiled, not run on hardware
//
#include <hip/hip_runtime.h>
#include <hip/hip_bf16.h>
#include <stdint.h>

// ---------------------------------------------------------------------------
// TransitionUp fused pipeline for MI455X (gfx1250, wave32, WMMA).
// Main GEMM y = x @ W1[:, :256]^T done with v_wmma_f32_16x16x32_bf16 using a
// 3-term bf16 hi/lo split for fp32-grade accuracy at bf16 matrix throughput.
// BN statistics folded algebraically so x is read exactly once.
// ---------------------------------------------------------------------------

#define TC 256          // channels
#define NPTS 262144     // points
#define NSEG 8
#define SEGSZ 32768
#define BR 32           // rows per chunk
#define CHUNKS 8        // chunks per block -> 256 rows / block
#define BLK_ROWS (BR*CHUNKS)
#define NBLK (NPTS/BLK_ROWS)        // 1024
#define BLKS_PER_SEG (SEGSZ/BLK_ROWS) // 128

typedef __attribute__((ext_vector_type(16))) __bf16 v16bf;
typedef __attribute__((ext_vector_type(8)))  float  v8f;

union Frag { uint4 q[2]; v16bf v; };

__device__ __forceinline__ unsigned short f32_to_bf16_rne(float f) {
    unsigned int u = __builtin_bit_cast(unsigned int, f);
    unsigned int r = 0x7FFFu + ((u >> 16) & 1u);
    return (unsigned short)((u + r) >> 16);
}
__device__ __forceinline__ float bf16u_to_f32(unsigned short h) {
    unsigned int u = ((unsigned int)h) << 16;
    return __builtin_bit_cast(float, u);
}

// ---------------------------------------------------------------------------
// K0: split W1a = W1[:, :256] into bf16 hi/lo, slab order [ks][c][k_local].
// ---------------------------------------------------------------------------
__global__ void k0_split_w(const float* __restrict__ W1,
                           unsigned short* __restrict__ Wh,
                           unsigned short* __restrict__ Wl) {
    int c = blockIdx.x;      // 0..255 (output channel)
    int k = threadIdx.x;     // 0..255 (reduction index)
    float f = W1[(size_t)c * 512 + k];
    unsigned short hb = f32_to_bf16_rne(f);
    unsigned short lb = f32_to_bf16_rne(f - bf16u_to_f32(hb));
    size_t o = (size_t)(k >> 5) * (TC * 32) + (size_t)c * 32 + (k & 31);
    Wh[o] = hb;
    Wl[o] = lb;
}

// ---------------------------------------------------------------------------
// K3: main GEMM  y[i][c] = sum_k x[i][k] * W1[c][k]  via bf16-split WMMA.
// Also emits per-block partials: col-sums of x, col-sums of y, col-sumsq of y.
// Block = 256 threads = 8 waves (2 row-groups x 4 col-groups), 8 chunks of
// 32 rows. Dynamic LDS 64KB: x hi/lo (32KB) + W slab hi/lo (32KB, reused as
// reduction scratch after the K loop).
// ---------------------------------------------------------------------------
__global__ __launch_bounds__(256) void k3_gemm(
    const float* __restrict__ x,
    const unsigned short* __restrict__ Wh,
    const unsigned short* __restrict__ Wl,
    float* __restrict__ y,
    float* __restrict__ p_xsum,
    float* __restrict__ p_ysum,
    float* __restrict__ p_ysq)
{
    extern __shared__ char smem[];
    unsigned short* xh = (unsigned short*)smem;     // 32*256 = 16KB
    unsigned short* xl = xh + BR * TC;              // 16KB
    unsigned short* wh = xl + BR * TC;              // 256*32 = 16KB
    unsigned short* wl = wh + TC * 32;              // 16KB
    float* red = (float*)wh;                        // alias: 2 x [4][256] f32

    const int t    = threadIdx.x;
    const int lane = t & 31;
    const int wave = t >> 5;
    const int rg   = wave & 1;        // row group (16 rows each)
    const int cg   = wave >> 1;       // col group (64 channels each)
    const int lh   = lane & 15;
    const int sel  = lane >> 4;       // lane half (K-half selector)
    const int arow = rg * 16 + lh;    // local row this lane owns in A
    const int akey = arow & 15;       // LDS granule swizzle key for x

    float xs_acc = 0.f, ys_acc = 0.f, yq_acc = 0.f;
    const size_t row_base0 = (size_t)blockIdx.x * BLK_ROWS;

    for (int chunk = 0; chunk < CHUNKS; ++chunk) {
        const size_t row0 = row_base0 + (size_t)chunk * BR;

        // ---- stage x tile as bf16 hi/lo (swizzled 16B granules) ----------
        __syncthreads();
        for (int r = 0; r < BR; ++r) {
            float f = x[(row0 + r) * TC + t];        // coalesced
            xs_acc += f;                              // exact fp32 col-sum
            unsigned short hb = f32_to_bf16_rne(f);
            unsigned short lb = f32_to_bf16_rne(f - bf16u_to_f32(hb));
            int g   = t >> 3;                         // granule 0..31
            int off = r * TC + (((g ^ (r & 15)) << 3) | (t & 7));
            xh[off] = hb;
            xl[off] = lb;
        }

        v8f acc0 = {}, acc1 = {}, acc2 = {}, acc3 = {};

        for (int ks = 0; ks < 8; ++ks) {
            __syncthreads();
            // ---- stage 32x256 W slab (hi/lo), granule-swizzled -----------
            {
                const uint4* sH = (const uint4*)(Wh + (size_t)ks * (TC * 32));
                const uint4* sL = (const uint4*)(Wl + (size_t)ks * (TC * 32));
                uint4* dH = (uint4*)wh;
                uint4* dL = (uint4*)wl;
#pragma unroll
                for (int i = 0; i < 4; ++i) {
                    int gf = t * 4 + i;               // 0..1023 granules
                    int c  = gf >> 2;
                    int g  = gf & 3;
                    int d  = c * 4 + (g ^ ((c >> 2) & 3));
                    dH[d] = sH[gf];
                    dL[d] = sL[gf];
                }
            }
            __syncthreads();

            // ---- A fragments (16x32 bf16 layout per CDNA5 ISA 7.12.2) ----
            Frag Ah, Al;
            {
                const uint4* xh4 = (const uint4*)xh;
                const uint4* xl4 = (const uint4*)xl;
                int g0 = (ks * 4 + sel)     ^ akey;   // K = ks*32 + sel*8 ..
                int g1 = (ks * 4 + 2 + sel) ^ akey;   // K = ks*32+16+sel*8 ..
                Ah.q[0] = xh4[arow * 32 + g0];
                Ah.q[1] = xh4[arow * 32 + g1];
                Al.q[0] = xl4[arow * 32 + g0];
                Al.q[1] = xl4[arow * 32 + g1];
            }

            // ---- B fragments + 3-term WMMA per 16-wide channel tile ------
            const uint4* wh4 = (const uint4*)wh;
            const uint4* wl4 = (const uint4*)wl;
#pragma unroll
            for (int nt = 0; nt < 4; ++nt) {
                int c  = cg * 64 + nt * 16 + lh;      // B column (channel)
                int bk = (c >> 2) & 3;
                Frag Bh, Bl;
                Bh.q[0] = wh4[c * 4 + ((sel)     ^ bk)];
                Bh.q[1] = wh4[c * 4 + ((sel + 2) ^ bk)];
                Bl.q[0] = wl4[c * 4 + ((sel)     ^ bk)];
                Bl.q[1] = wl4[c * 4 + ((sel + 2) ^ bk)];
                v8f* ap = (nt == 0) ? &acc0 : (nt == 1) ? &acc1
                        : (nt == 2) ? &acc2 : &acc3;
                *ap = __builtin_amdgcn_wmma_f32_16x16x32_bf16(
                          false, Ah.v, false, Bh.v, (short)0, *ap, false, false);
                *ap = __builtin_amdgcn_wmma_f32_16x16x32_bf16(
                          false, Al.v, false, Bh.v, (short)0, *ap, false, false);
                *ap = __builtin_amdgcn_wmma_f32_16x16x32_bf16(
                          false, Ah.v, false, Bl.v, (short)0, *ap, false, false);
            }
        }

        // ---- write y and form per-channel partial sums -------------------
        float s[4], q[4];
#pragma unroll
        for (int nt = 0; nt < 4; ++nt) {
            v8f a = (nt == 0) ? acc0 : (nt == 1) ? acc1
                  : (nt == 2) ? acc2 : acc3;
            float ls = 0.f, lq = 0.f;
#pragma unroll
            for (int v = 0; v < 8; ++v) {
                float val = a[v];
                size_t rr = row0 + (size_t)(rg * 16 + v + 8 * sel);
                y[rr * TC + cg * 64 + nt * 16 + lh] = val;
                ls += val;
                lq += val * val;
            }
            s[nt] = ls; q[nt] = lq;
        }

        __syncthreads();          // slab no longer needed -> reuse as scratch
        {
            int contrib = rg * 2 + sel;               // 4 contributors / chan
#pragma unroll
            for (int nt = 0; nt < 4; ++nt) {
                int c = cg * 64 + nt * 16 + lh;
                red[contrib * 256 + c]        = s[nt];
                red[1024 + contrib * 256 + c] = q[nt];
            }
        }
        __syncthreads();
        {   // fixed-order combine -> deterministic
            ys_acc += ((red[t] + red[256 + t]) + (red[512 + t] + red[768 + t]));
            yq_acc += ((red[1024 + t] + red[1280 + t]) +
                       (red[1536 + t] + red[1792 + t]));
        }
    }

    p_xsum[(size_t)blockIdx.x * TC + t] = xs_acc;
    p_ysum[(size_t)blockIdx.x * TC + t] = ys_acc;
    p_ysq [(size_t)blockIdx.x * TC + t] = yq_acc;
}

// ---------------------------------------------------------------------------
// K4: deterministic fixed-order reduction of block partials per segment.
// ---------------------------------------------------------------------------
__global__ void k4_reduce(const float* __restrict__ p_xsum,
                          const float* __restrict__ p_ysum,
                          const float* __restrict__ p_ysq,
                          float* __restrict__ seg_xsum,
                          float* __restrict__ seg_ysum,
                          float* __restrict__ seg_ysq) {
    int seg = blockIdx.x;     // 0..7
    int c   = threadIdx.x;    // 0..255
    float sx = 0.f, sy = 0.f, sq = 0.f;
    for (int b = 0; b < BLKS_PER_SEG; ++b) {
        size_t off = ((size_t)(seg * BLKS_PER_SEG + b)) * TC + c;
        sx += p_xsum[off];
        sy += p_ysum[off];
        sq += p_ysq[off];
    }
    seg_xsum[seg * TC + c] = sx;
    seg_ysum[seg * TC + c] = sy;
    seg_ysq [seg * TC + c] = sq;
}

// ---------------------------------------------------------------------------
// K2: tiny prep (1 block). g = relu(mean @ W2^T + b2); h = g @ W1b^T + b1;
// exact BN stats via  z = y + h[seg]; emit A[c] and T[b][c] so that
// out = relu(A[c]*y + T[seg][c]).
// ---------------------------------------------------------------------------
__global__ void k2_prep(const float* __restrict__ W1,
                        const float* __restrict__ b1,
                        const float* __restrict__ gamma,
                        const float* __restrict__ beta,
                        const float* __restrict__ W2,
                        const float* __restrict__ b2,
                        const float* __restrict__ seg_xsum,
                        const float* __restrict__ seg_ysum,
                        const float* __restrict__ seg_ysq,
                        float* __restrict__ Aarr,
                        float* __restrict__ Tarr) {
    __shared__ float sm[NSEG][TC];
    __shared__ float gg[NSEG][TC];
    int c = threadIdx.x;
    for (int b = 0; b < NSEG; ++b)
        sm[b][c] = seg_xsum[b * TC + c] * (1.0f / (float)SEGSZ);
    __syncthreads();
    for (int b = 0; b < NSEG; ++b) {
        float acc = b2[c];
        for (int k = 0; k < TC; ++k)
            acc = fmaf(sm[b][k], W2[(size_t)c * TC + k], acc);
        gg[b][c] = fmaxf(acc, 0.0f);
    }
    __syncthreads();
    float hv[NSEG];
    for (int b = 0; b < NSEG; ++b) {
        float acc = b1[c];
        for (int k = 0; k < TC; ++k)
            acc = fmaf(gg[b][k], W1[(size_t)c * 512 + 256 + k], acc);
        hv[b] = acc;
    }
    float msum = 0.f, qsum = 0.f;
    for (int b = 0; b < NSEG; ++b) {
        float ys = seg_ysum[b * TC + c];
        msum += ys + (float)SEGSZ * hv[b];
        qsum += seg_ysq[b * TC + c] + 2.0f * hv[b] * ys
              + (float)SEGSZ * hv[b] * hv[b];
    }
    const float invN = 1.0f / (float)NPTS;
    float mu  = msum * invN;
    float var = qsum * invN - mu * mu;
    float inv = rsqrtf(var + 1e-5f);
    float a   = gamma[c] * inv;
    Aarr[c] = a;
    for (int b = 0; b < NSEG; ++b)
        Tarr[b * TC + c] = (hv[b] - mu) * a + beta[c];
}

// ---------------------------------------------------------------------------
// K5: in-place finalize  out = relu(A[c]*y + T[seg][c]),  float4 vectorized.
// ---------------------------------------------------------------------------
__global__ __launch_bounds__(256) void k5_final(float* __restrict__ y,
                                                const float* __restrict__ Aarr,
                                                const float* __restrict__ Tarr) {
    size_t idx = (size_t)blockIdx.x * blockDim.x + threadIdx.x; // one float4
    size_t f   = idx * 4;
    int    c0  = (int)(f & (TC - 1));
    size_t row = f >> 8;
    int    seg = (int)(row >> 15);
    float4 v = *(const float4*)(y + f);
    float4 a = *(const float4*)(Aarr + c0);
    float4 tt = *(const float4*)(Tarr + (size_t)seg * TC + c0);
    v.x = fmaxf(fmaf(v.x, a.x, tt.x), 0.f);
    v.y = fmaxf(fmaf(v.y, a.y, tt.y), 0.f);
    v.z = fmaxf(fmaf(v.z, a.z, tt.z), 0.f);
    v.w = fmaxf(fmaf(v.w, a.w, tt.w), 0.f);
    *(float4*)(y + f) = v;
}

// ---------------------------------------------------------------------------
extern "C" void kernel_launch(void* const* d_in, const int* in_sizes, int n_in,
                              void* d_out, int out_size, void* d_ws, size_t ws_size,
                              hipStream_t stream) {
    (void)in_sizes; (void)n_in; (void)out_size; (void)ws_size;
    // setup_inputs order: p, x, o, W1, b1, gamma1, beta1, W2, b2
    const float* x     = (const float*)d_in[1];
    const float* W1    = (const float*)d_in[3];
    const float* b1    = (const float*)d_in[4];
    const float* gamma = (const float*)d_in[5];
    const float* beta  = (const float*)d_in[6];
    const float* W2    = (const float*)d_in[7];
    const float* b2    = (const float*)d_in[8];
    float* out = (float*)d_out;

    // workspace carve-up (bytes)
    char* ws = (char*)d_ws;
    unsigned short* Wh = (unsigned short*)(ws);                    // 128KB
    unsigned short* Wl = (unsigned short*)(ws + (128 << 10));      // 128KB
    float* p_xsum   = (float*)(ws + (256 << 10));                  // 1MB
    float* p_ysum   = (float*)(ws + (256 << 10) + (NBLK * TC * 4));
    float* p_ysq    = (float*)(ws + (256 << 10) + 2 * (NBLK * TC * 4));
    char*  tail     = ws + (256 << 10) + 3 * (NBLK * TC * 4);
    float* seg_xsum = (float*)(tail);
    float* seg_ysum = (float*)(tail + NSEG * TC * 4);
    float* seg_ysq  = (float*)(tail + 2 * NSEG * TC * 4);
    float* Aarr     = (float*)(tail + 3 * NSEG * TC * 4);
    float* Tarr     = (float*)(tail + 3 * NSEG * TC * 4 + TC * 4);

    k0_split_w<<<TC, TC, 0, stream>>>(W1, Wh, Wl);
    k3_gemm<<<NBLK, 256, 64 * 1024, stream>>>(x, Wh, Wl, out,
                                              p_xsum, p_ysum, p_ysq);
    k4_reduce<<<NSEG, TC, 0, stream>>>(p_xsum, p_ysum, p_ysq,
                                       seg_xsum, seg_ysum, seg_ysq);
    k2_prep<<<1, TC, 0, stream>>>(W1, b1, gamma, beta, W2, b2,
                                  seg_xsum, seg_ysum, seg_ysq, Aarr, Tarr);
    k5_final<<<(NPTS * TC / 4) / 256, 256, 0, stream>>>(out, Aarr, Tarr);
}
